// Per_Region_Encoding_76381698392783
// MI455X (gfx1250) — matrix operand order repaired and checked by reading, not compiled, not run on hardware
//
#include <hip/hip_runtime.h>
#include <hip/hip_bf16.h>

// ---------------------------------------------------------------------------
// MI455X (gfx1250) implementation: bf16 WMMA implicit-GEMM convolutions.
// v_wmma_f32_16x16x32_bf16 (wave32, f32 accum). Activations NHWC bf16 so each
// K-chunk of 32 channels is two contiguous 16B loads per lane (ISA A-fragment
// layout). Weights pre-packed device-side into per-lane fragment order (one
// 32B load per lane per B tile). Each wave computes 2 M-tiles x 4 N-tiles:
// 8 WMMAs per 4 B-tile loads; loads are packed ahead of the WMMA group so the
// matrix pipe overlaps the next k-step's memory traffic.
// ---------------------------------------------------------------------------

typedef __attribute__((ext_vector_type(16))) __bf16 v16bf;
typedef __attribute__((ext_vector_type(8)))  float  v8f;

#define WMMA_BF16(a, b, c) \
    __builtin_amdgcn_wmma_f32_16x16x32_bf16(false, (a), false, (b), (short)0, (c), false, false)

static __device__ __forceinline__ v8f zero_v8f() {
    v8f z = {0.f, 0.f, 0.f, 0.f, 0.f, 0.f, 0.f, 0.f};
    return z;
}

union AFrag { v16bf v; uint4 q[2]; };

// ------------------------- layout / transpose helpers -----------------------

// x: [4,256,32,32] fp32 NCHW  ->  xb: [4,32,32,256] bf16 NHWC
__global__ __launch_bounds__(256) void nchw2nhwc_kernel(const float* __restrict__ in,
                                                        __bf16* __restrict__ out) {
    const int HW = 1024, C = 256;
    int idx = blockIdx.x * 256 + threadIdx.x;      // 4*256*1024 threads
    int n = idx / (C * HW);
    int r = idx % (C * HW);
    int c = r / HW;
    int p = r % HW;
    out[((size_t)n * HW + p) * C + c] = (__bf16)in[idx];
}

// InstanceNorm over NHWC bf16: one workgroup per (n, c). Population variance.
__global__ __launch_bounds__(256) void inorm_kernel(const __bf16* __restrict__ in,
                                                    __bf16* __restrict__ out, int P) {
    __shared__ float s1[256];
    __shared__ float s2[256];
    const int C = 256;
    int c = blockIdx.x & 255;
    int n = blockIdx.x >> 8;
    const __bf16* base = in + (size_t)n * P * C + c;
    float sum = 0.f, sq = 0.f;
    for (int p = threadIdx.x; p < P; p += 256) {
        float v = (float)base[(size_t)p * C];
        sum += v; sq += v * v;
    }
    int t = threadIdx.x;
    s1[t] = sum; s2[t] = sq;
    __syncthreads();
    for (int off = 128; off > 0; off >>= 1) {
        if (t < off) { s1[t] += s1[t + off]; s2[t] += s2[t + off]; }
        __syncthreads();
    }
    float mean = s1[0] / (float)P;
    float var  = s2[0] / (float)P - mean * mean;
    float inv  = rsqrtf(var + 1e-5f);
    __bf16* ob = out + (size_t)n * P * C + c;
    for (int p = threadIdx.x; p < P; p += 256) {
        float v = (float)base[(size_t)p * C];
        ob[(size_t)p * C] = (__bf16)((v - mean) * inv);
    }
}

// ------------------------- weight fragment packing --------------------------
// Fragment element order: frag[kb][nb][lane][e], 16 bf16 per lane per tile.
// lane<16: col n = nb*16+lane, K = kb*32 + (e<8 ? e : 16 + e-8)
// lane>=16: col n = nb*16+lane-16, K = kb*32 + (e<8 ? 8 + e : 24 + e-8)

// 3x3 conv (OIHW [256,256,3,3]) or 1x1 (is1x1=1): K = tap*256 + cin.
__global__ __launch_bounds__(256) void repack_conv_b(const float* __restrict__ w,
                                                     __bf16* __restrict__ frag,
                                                     int KB, int is1x1) {
    int tid = blockIdx.x * 256 + threadIdx.x;
    if (tid >= KB * 8192) return;
    int e    = tid & 15;
    int lane = (tid >> 4) & 31;
    int nb   = (tid >> 9) & 15;
    int kb   = tid >> 13;
    int n   = nb * 16 + (lane & 15);
    int k   = kb * 32 + ((lane < 16) ? 0 : 8) + ((e < 8) ? e : 8 + e);
    int cin = k & 255;
    int tap = k >> 8;
    int ky = tap / 3, kx = tap % 3;
    size_t idx = is1x1 ? ((size_t)n * 256 + cin)
                       : ((((size_t)n * 256 + cin) * 3 + ky) * 3 + kx);
    frag[tid] = (__bf16)w[idx];
}

// Fused convT (main + shortcut) per output-parity class.
// K enumeration: k = ((src*nt + t) * 256) + cin, tap t -> (ky in Ty, kx in Tx).
// Torch ConvTranspose weight layout: [Cin, Cout, 3, 3], gather form (no flip).
__global__ __launch_bounds__(256) void repack_convT_b(const float* __restrict__ wdp,
                                                      const float* __restrict__ wsp,
                                                      __bf16* __restrict__ frag,
                                                      int KB, int nt, int ntx,
                                                      int ty0, int ty1, int tx0, int tx1) {
    int tid = blockIdx.x * 256 + threadIdx.x;
    if (tid >= KB * 8192) return;
    int e    = tid & 15;
    int lane = (tid >> 4) & 31;
    int nb   = (tid >> 9) & 15;
    int kb   = tid >> 13;
    int n   = nb * 16 + (lane & 15);
    int k   = kb * 32 + ((lane < 16) ? 0 : 8) + ((e < 8) ? e : 8 + e);
    int cin = k & 255;
    int tt  = k >> 8;                 // 0 .. 2*nt-1
    int s   = (tt >= nt) ? 1 : 0;
    int t   = tt - s * nt;
    int ky  = ((t / ntx) == 0) ? ty0 : ty1;
    int kx  = ((t % ntx) == 0) ? tx0 : tx1;
    const float* wsrc = s ? wsp : wdp;
    frag[tid] = (__bf16)wsrc[(((size_t)cin * 256 + n) * 3 + ky) * 3 + kx];
}

// ------------------------- WMMA GEMM conv kernels ---------------------------
// 128 threads = 4 waves. Block M-tile = 32 pixels (2 WMMA M tiles); wave w
// computes couts [w*64, w*64+64). Per k-step: 4 A loads + 4 B-tile loads are
// issued as one pack, then 8 WMMAs run back-to-back.

__global__ __launch_bounds__(128) void conv3x3_wmma_kernel(
        const __bf16* __restrict__ A, const __bf16* __restrict__ Bf,
        const float* __restrict__ bias, __bf16* __restrict__ out, int H, int W) {
    const int C = 256;
    int lane = threadIdx.x & 31;
    int wave = threadIdx.x >> 5;
    int m    = lane & 15;
    int pidB = blockIdx.x * 32;
    int pid0 = pidB + m;
    int pid1 = pidB + 16 + m;
    int HW = H * W;
    int n0 = pid0 / HW, r0 = pid0 % HW, y0 = r0 / W, x0 = r0 % W;
    int n1 = pid1 / HW, r1 = pid1 % HW, y1 = r1 / W, x1 = r1 % W;
    int coff = (lane < 16) ? 0 : 8;

    v8f acc0[4], acc1[4];
    #pragma unroll
    for (int j = 0; j < 4; ++j) { acc0[j] = zero_v8f(); acc1[j] = zero_v8f(); }

    const v16bf* bp = (const v16bf*)Bf + ((size_t)wave * 4) * 32 + lane;

    #pragma unroll 1
    for (int kb = 0; kb < 72; ++kb) {
        int tap  = kb >> 3;
        int cinb = (kb & 7) << 5;
        int dy = tap / 3 - 1, dx = tap % 3 - 1;

        AFrag a0, a1;
        int iy0 = y0 + dy, ix0 = x0 + dx;
        if ((unsigned)iy0 < (unsigned)H && (unsigned)ix0 < (unsigned)W) {
            const __bf16* ap = A + (((size_t)(n0 * H + iy0) * W + ix0) * C + cinb + coff);
            a0.q[0] = *(const uint4*)ap;
            a0.q[1] = *(const uint4*)(ap + 16);
        } else { a0.q[0] = make_uint4(0,0,0,0); a0.q[1] = make_uint4(0,0,0,0); }
        int iy1 = y1 + dy, ix1 = x1 + dx;
        if ((unsigned)iy1 < (unsigned)H && (unsigned)ix1 < (unsigned)W) {
            const __bf16* ap = A + (((size_t)(n1 * H + iy1) * W + ix1) * C + cinb + coff);
            a1.q[0] = *(const uint4*)ap;
            a1.q[1] = *(const uint4*)(ap + 16);
        } else { a1.q[0] = make_uint4(0,0,0,0); a1.q[1] = make_uint4(0,0,0,0); }

        v16bf b0 = bp[0], b1 = bp[32], b2 = bp[64], b3 = bp[96];
        bp += 512;                     // next kb (16 nb tiles * 32 lanes)

        acc0[0] = WMMA_BF16(a0.v, b0, acc0[0]);
        acc0[1] = WMMA_BF16(a0.v, b1, acc0[1]);
        acc0[2] = WMMA_BF16(a0.v, b2, acc0[2]);
        acc0[3] = WMMA_BF16(a0.v, b3, acc0[3]);
        acc1[0] = WMMA_BF16(a1.v, b0, acc1[0]);
        acc1[1] = WMMA_BF16(a1.v, b1, acc1[1]);
        acc1[2] = WMMA_BF16(a1.v, b2, acc1[2]);
        acc1[3] = WMMA_BF16(a1.v, b3, acc1[3]);
    }

    #pragma unroll
    for (int j = 0; j < 4; ++j) {
        int cout = wave * 64 + j * 16 + (lane & 15);
        float bv = bias[cout];
        #pragma unroll
        for (int e = 0; e < 8; ++e) {
            int mm = e + ((lane < 16) ? 0 : 8);
            float v = acc0[j][e] + bv;
            v = (v >= 0.f) ? v : 0.1f * v;               // LeakyReLU(0.1)
            out[(size_t)(pidB + mm) * C + cout] = (__bf16)v;
            float v2 = acc1[j][e] + bv;
            v2 = (v2 >= 0.f) ? v2 : 0.1f * v2;
            out[(size_t)(pidB + 16 + mm) * C + cout] = (__bf16)v2;
        }
    }
}

// Fused ConvTranspose(main) + ConvTranspose(shortcut) for one output parity.
// Output pixel (oy,ox) = (2*ys+py, 2*xs+px); gather taps iy=(oy+1-ky)/2.
__global__ __launch_bounds__(128) void convT_wmma_kernel(
        const __bf16* __restrict__ Amain, const __bf16* __restrict__ Ashort,
        const __bf16* __restrict__ Bf, const float* __restrict__ bd,
        const float* __restrict__ bs, __bf16* __restrict__ out,
        int H, int W, int py, int px, int nt, int ntx,
        int ty0, int ty1, int tx0, int tx1) {
    const int C = 256;
    int lane = threadIdx.x & 31;
    int wave = threadIdx.x >> 5;
    int m    = lane & 15;
    int pidB = blockIdx.x * 32;
    int pid0 = pidB + m;
    int pid1 = pidB + 16 + m;
    int HW = H * W;
    int n0 = pid0 / HW, r0 = pid0 % HW;
    int oy0 = 2 * (r0 / W) + py, ox0 = 2 * (r0 % W) + px;
    int n1 = pid1 / HW, r1 = pid1 % HW;
    int oy1 = 2 * (r1 / W) + py, ox1 = 2 * (r1 % W) + px;
    int coff = (lane < 16) ? 0 : 8;
    int KB = 16 * nt;                                   // K = 2 srcs * nt taps * 256

    v8f acc0[4], acc1[4];
    #pragma unroll
    for (int j = 0; j < 4; ++j) { acc0[j] = zero_v8f(); acc1[j] = zero_v8f(); }

    const v16bf* bp = (const v16bf*)Bf + ((size_t)wave * 4) * 32 + lane;

    #pragma unroll 1
    for (int kb = 0; kb < KB; ++kb) {
        int cinb = (kb & 7) << 5;
        int tt = kb >> 3;
        int s  = (tt >= nt) ? 1 : 0;
        int t  = tt - s * nt;
        int ky = ((t / ntx) == 0) ? ty0 : ty1;
        int kx = ((t % ntx) == 0) ? tx0 : tx1;
        const __bf16* src = s ? Ashort : Amain;

        AFrag a0, a1;
        int iy0 = (oy0 + 1 - ky) >> 1, ix0 = (ox0 + 1 - kx) >> 1;
        if ((unsigned)iy0 < (unsigned)H && (unsigned)ix0 < (unsigned)W) {
            const __bf16* ap = src + (((size_t)(n0 * H + iy0) * W + ix0) * C + cinb + coff);
            a0.q[0] = *(const uint4*)ap;
            a0.q[1] = *(const uint4*)(ap + 16);
        } else { a0.q[0] = make_uint4(0,0,0,0); a0.q[1] = make_uint4(0,0,0,0); }
        int iy1 = (oy1 + 1 - ky) >> 1, ix1 = (ox1 + 1 - kx) >> 1;
        if ((unsigned)iy1 < (unsigned)H && (unsigned)ix1 < (unsigned)W) {
            const __bf16* ap = src + (((size_t)(n1 * H + iy1) * W + ix1) * C + cinb + coff);
            a1.q[0] = *(const uint4*)ap;
            a1.q[1] = *(const uint4*)(ap + 16);
        } else { a1.q[0] = make_uint4(0,0,0,0); a1.q[1] = make_uint4(0,0,0,0); }

        v16bf b0 = bp[0], b1 = bp[32], b2 = bp[64], b3 = bp[96];
        bp += 512;

        acc0[0] = WMMA_BF16(a0.v, b0, acc0[0]);
        acc0[1] = WMMA_BF16(a0.v, b1, acc0[1]);
        acc0[2] = WMMA_BF16(a0.v, b2, acc0[2]);
        acc0[3] = WMMA_BF16(a0.v, b3, acc0[3]);
        acc1[0] = WMMA_BF16(a1.v, b0, acc1[0]);
        acc1[1] = WMMA_BF16(a1.v, b1, acc1[1]);
        acc1[2] = WMMA_BF16(a1.v, b2, acc1[2]);
        acc1[3] = WMMA_BF16(a1.v, b3, acc1[3]);
    }

    int Ho = 2 * H, Wo = 2 * W;
    #pragma unroll
    for (int j = 0; j < 4; ++j) {
        int cout = wave * 64 + j * 16 + (lane & 15);
        float bb = bd[cout] + bs[cout];
        #pragma unroll
        for (int e = 0; e < 8; ++e) {
            int mm = e + ((lane < 16) ? 0 : 8);
            int p2 = pidB + mm;
            int n2 = p2 / HW, r2 = p2 % HW;
            int oy2 = 2 * (r2 / W) + py, ox2 = 2 * (r2 % W) + px;
            out[(((size_t)n2 * Ho + oy2) * Wo + ox2) * C + cout] = (__bf16)(acc0[j][e] + bb);
            int p3 = pidB + 16 + mm;
            int n3 = p3 / HW, r3 = p3 % HW;
            int oy3 = 2 * (r3 / W) + py, ox3 = 2 * (r3 % W) + px;
            out[(((size_t)n3 * Ho + oy3) * Wo + ox3) * C + cout] = (__bf16)(acc1[j][e] + bb);
        }
    }
}

// Final 1x1 conv + bias + tanh, NHWC bf16 -> NCHW fp32 (written into d_out).
__global__ __launch_bounds__(128) void conv1x1_tanh_kernel(
        const __bf16* __restrict__ A, const __bf16* __restrict__ Bf,
        const float* __restrict__ bias, float* __restrict__ out) {
    const int C = 256, H = 256, W = 256, HW = 65536;
    int lane = threadIdx.x & 31;
    int wave = threadIdx.x >> 5;
    int m    = lane & 15;
    int pid  = blockIdx.x * 16 + m;
    int coff = (lane < 16) ? 0 : 8;

    v8f acc[4];
    #pragma unroll
    for (int j = 0; j < 4; ++j) acc[j] = zero_v8f();

    const v16bf* bp = (const v16bf*)Bf + ((size_t)wave * 4) * 32 + lane;

    #pragma unroll 1
    for (int kb = 0; kb < 8; ++kb) {
        int cinb = kb << 5;
        const __bf16* ap = A + ((size_t)pid * C + cinb + coff);
        AFrag au;
        au.q[0] = *(const uint4*)ap;
        au.q[1] = *(const uint4*)(ap + 16);
        v16bf b0 = bp[0], b1 = bp[32], b2 = bp[64], b3 = bp[96];
        bp += 512;
        acc[0] = WMMA_BF16(au.v, b0, acc[0]);
        acc[1] = WMMA_BF16(au.v, b1, acc[1]);
        acc[2] = WMMA_BF16(au.v, b2, acc[2]);
        acc[3] = WMMA_BF16(au.v, b3, acc[3]);
    }

    #pragma unroll
    for (int j = 0; j < 4; ++j) {
        int cout = wave * 64 + j * 16 + (lane & 15);
        float bv = bias[cout];
        #pragma unroll
        for (int e = 0; e < 8; ++e) {
            int mm = e + ((lane < 16) ? 0 : 8);
            int p2 = blockIdx.x * 16 + mm;
            int n2 = p2 / HW;
            int r2 = p2 % HW;
            int y2 = r2 / W, x2 = r2 % W;
            out[(((size_t)n2 * C + cout) * H + y2) * W + x2] = tanhf(acc[j][e] + bv);
        }
    }
}

// ------------------------- segment pooling ----------------------------------
// seg is one-hot over 8 labels (a partition of all pixels); h is 256x256 so
// the nearest-resize index map is the identity. Per-(b,row) LDS histogram.
__global__ __launch_bounds__(256) void seg_pool_kernel(const float* __restrict__ seg,
                                                       const float* __restrict__ h,
                                                       float* __restrict__ gsums,
                                                       float* __restrict__ gcnt) {
    __shared__ float sums[2048];   // [8 segs][256 ch]
    __shared__ float cnt[8];
    int b = blockIdx.x >> 8;
    int y = blockIdx.x & 255;
    int t = threadIdx.x;           // pixel x
    for (int i = t; i < 2048; i += 256) sums[i] = 0.f;
    if (t < 8) cnt[t] = 0.f;
    __syncthreads();

    int l = 0;
    #pragma unroll
    for (int s = 0; s < 8; ++s)
        if (seg[(((size_t)b * 8 + s) * 256 + y) * 256 + t] > 0.f) l = s;
    atomicAdd(&cnt[l], 1.f);

    for (int c = 0; c < 256; ++c) {
        float v = h[(((size_t)b * 256 + c) * 256 + y) * 256 + t];
        atomicAdd(&sums[l * 256 + c], v);
    }
    __syncthreads();

    for (int i = t; i < 2048; i += 256) atomicAdd(&gsums[(size_t)b * 2048 + i], sums[i]);
    if (t < 8) atomicAdd(&gcnt[b * 8 + t], cnt[t]);
}

__global__ __launch_bounds__(256) void finalize_kernel(const float* __restrict__ gsums,
                                                       const float* __restrict__ gcnt,
                                                       float* __restrict__ outp) {
    int tid = blockIdx.x * 256 + threadIdx.x;    // 0..9215 -> codes[b][s][c]
    int b = tid / 2304;
    int r = tid % 2304;
    int s = r >> 8;
    int c = r & 255;
    float val;
    if (s < 8) {
        float cn = gcnt[b * 8 + s];
        val = (cn > 0.f) ? gsums[((size_t)b * 8 + s) * 256 + c] / cn : 0.f;
    } else {   // global mean = sum over the segment partition / 65536
        float acc = 0.f;
        #pragma unroll
        for (int s2 = 0; s2 < 8; ++s2) acc += gsums[((size_t)b * 8 + s2) * 256 + c];
        val = acc * (1.0f / 65536.0f);
    }
    outp[tid] = val;
    if (s == 0 && c < 8)
        outp[9216 + b * 8 + c] = (gcnt[b * 8 + c] > 0.f) ? 1.f : 0.f;
}

// ------------------------- host orchestration -------------------------------

extern "C" void kernel_launch(void* const* d_in, const int* in_sizes, int n_in,
                              void* d_out, int out_size, void* d_ws, size_t ws_size,
                              hipStream_t stream) {
    (void)in_sizes; (void)n_in; (void)out_size; (void)ws_size;

    const float* x   = (const float*)d_in[0];
    const float* seg = (const float*)d_in[1];
    const float* wc[3] = {(const float*)d_in[2],  (const float*)d_in[8],  (const float*)d_in[14]};
    const float* bc[3] = {(const float*)d_in[3],  (const float*)d_in[9],  (const float*)d_in[15]};
    const float* wd[3] = {(const float*)d_in[4],  (const float*)d_in[10], (const float*)d_in[16]};
    const float* bd[3] = {(const float*)d_in[5],  (const float*)d_in[11], (const float*)d_in[17]};
    const float* wsc[3]= {(const float*)d_in[6],  (const float*)d_in[12], (const float*)d_in[18]};
    const float* bsc[3]= {(const float*)d_in[7],  (const float*)d_in[13], (const float*)d_in[19]};
    const float* wo = (const float*)d_in[20];
    const float* bo = (const float*)d_in[21];

    // Workspace bump allocator (256B aligned).
    char* base = (char*)d_ws;
    size_t off = 0;
    auto alloc = [&](size_t bytes) -> char* {
        char* p = base + off;
        off += (bytes + 255) & ~(size_t)255;
        return p;
    };

    __bf16* xb   = (__bf16*)alloc((size_t)4 * 1024  * 256 * 2);   // x NHWC bf16
    __bf16* nrmA = (__bf16*)alloc((size_t)4 * 16384 * 256 * 2);   // inorm(block in)
    __bf16* cvO  = (__bf16*)alloc((size_t)4 * 16384 * 256 * 2);   // conv3x3 + lrelu out
    __bf16* nrm2 = (__bf16*)alloc((size_t)4 * 16384 * 256 * 2);   // inorm(conv out)
    __bf16* h1   = (__bf16*)alloc((size_t)4 * 4096  * 256 * 2);
    __bf16* h2   = (__bf16*)alloc((size_t)4 * 16384 * 256 * 2);
    __bf16* h3   = (__bf16*)alloc((size_t)4 * 65536 * 256 * 2);
    __bf16* wcB[3];
    for (int i = 0; i < 3; ++i) wcB[i] = (__bf16*)alloc((size_t)72 * 8192 * 2);
    __bf16* wtB[3][4];
    for (int i = 0; i < 3; ++i)
        for (int p = 0; p < 4; ++p) {
            int nt = ((p >> 1) ? 2 : 1) * ((p & 1) ? 2 : 1);
            wtB[i][p] = (__bf16*)alloc((size_t)(16 * nt) * 8192 * 2);
        }
    __bf16* w1B = (__bf16*)alloc((size_t)8 * 8192 * 2);
    float* gsums = (float*)alloc((size_t)4 * 8 * 256 * 4);
    float* gcnt  = (float*)alloc((size_t)4 * 8 * 4);

    float* codes = (float*)d_out;                 // [4,9,256]
    float* hout  = (float*)d_out + 9248;          // [4,256,256,256] fp32 NCHW

    // ---- input transpose + weight packing ----
    nchw2nhwc_kernel<<<4096, 256, 0, stream>>>(x, xb);
    for (int i = 0; i < 3; ++i)
        repack_conv_b<<<(72 * 8192) / 256, 256, 0, stream>>>(wc[i], wcB[i], 72, 0);
    repack_conv_b<<<(8 * 8192) / 256, 256, 0, stream>>>(wo, w1B, 8, 1);
    for (int i = 0; i < 3; ++i)
        for (int p = 0; p < 4; ++p) {
            int py = p >> 1, px = p & 1;
            int nty = py ? 2 : 1, ntx = px ? 2 : 1;
            int nt = nty * ntx;
            int KB = 16 * nt;
            repack_convT_b<<<(KB * 8192) / 256, 256, 0, stream>>>(
                wd[i], wsc[i], wtB[i][p], KB, nt, ntx,
                py ? 0 : 1, 2, px ? 0 : 1, 2);
        }

    // ---- residual upsampling blocks ----
    const __bf16* bin = xb;
    __bf16* houts[3] = {h1, h2, h3};
    int H = 32, W = 32;
    for (int i = 0; i < 3; ++i) {
        int P = H * W;
        int tiles = 4 * P / 32;       // 32 output pixels per workgroup
        inorm_kernel<<<1024, 256, 0, stream>>>(bin, nrmA, P);
        conv3x3_wmma_kernel<<<tiles, 128, 0, stream>>>(nrmA, wcB[i], bc[i], cvO, H, W);
        inorm_kernel<<<1024, 256, 0, stream>>>(cvO, nrm2, P);
        for (int p = 0; p < 4; ++p) {
            int py = p >> 1, px = p & 1;
            int nty = py ? 2 : 1, ntx = px ? 2 : 1;
            int nt = nty * ntx;
            convT_wmma_kernel<<<tiles, 128, 0, stream>>>(
                nrm2, bin, wtB[i][p], bd[i], bsc[i], houts[i],
                H, W, py, px, nt, ntx, py ? 0 : 1, 2, px ? 0 : 1, 2);
        }
        bin = houts[i];
        H *= 2; W *= 2;
    }

    // ---- head: 1x1 conv + tanh -> h (fp32 NCHW in d_out) ----
    conv1x1_tanh_kernel<<<16384, 128, 0, stream>>>(h3, w1B, bo, hout);

    // ---- per-region pooling ----
    hipMemsetAsync(gsums, 0, (size_t)4 * 8 * 256 * 4, stream);
    hipMemsetAsync(gcnt,  0, (size_t)4 * 8 * 4, stream);
    seg_pool_kernel<<<1024, 256, 0, stream>>>(seg, hout, gsums, gcnt);
    finalize_kernel<<<36, 256, 0, stream>>>(gsums, gcnt, codes);
}
